// ResNet_28020366639553
// MI455X (gfx1250) — matrix operand order repaired and checked by reading, not compiled, not run on hardware
//
#include <hip/hip_runtime.h>

// ---------------------------------------------------------------------------
// Submanifold sparse-conv ResNet block for gfx1250 (MI455X).
// conv1(bf16 WMMA) -> BN+ReLU -> conv2(bf16 WMMA) -> BN -> +residual
// Activations for the GEMMs are bf16 (halves L2 gather traffic, enables
// v_wmma_f32_16x16x32_bf16); accumulation and BN are fp32.
// Each wave computes a 32x64 output tile (two 16x16x32 M-tiles x 4 Cout
// tiles) so every LDS B-fragment fetch feeds two WMMAs.
// ---------------------------------------------------------------------------

typedef __attribute__((ext_vector_type(16))) __bf16 v16bf;
typedef __attribute__((ext_vector_type(8)))  float  v8f;

#define CONV_K 9
#define CONV_C 64
// packed weight elements per conv: 9k * 2half * 4tile * 32lane * 16bf16
#define WPACK_ELEMS (9 * 2 * 4 * 32 * 16)   // 36864 bf16 = 73728 bytes
#define WPACK_BYTES (WPACK_ELEMS * 2)
#define LDS_BYTES   (WPACK_BYTES + 2 * 64 * 4)  // weights + sum/sumsq scratch

__device__ __forceinline__ unsigned short f2bf(float f) {
    unsigned int u = __float_as_uint(f);
    u += 0x7FFFu + ((u >> 16) & 1u);        // round-to-nearest-even
    return (unsigned short)(u >> 16);
}

// --- zero the 4 stats vectors (sum1, sq1, sum2, sq2 = 256 floats) -----------
__global__ void zero_stats_kernel(float* s) {
    if (threadIdx.x < 256) s[threadIdx.x] = 0.0f;
}

// --- pack W[k][cin][cout] (f32) into per-lane WMMA B fragments (bf16) -------
// element e = (((k*2+h)*4+t)*32+lane)*16+j
// B 32x16 (KxN): lane = half*16 + N; element j -> K = (j>>3)*16 + half*8 + (j&7) (within the
// 32-wide cin slice selected by h). cout = t*16 + (lane&15).
__global__ void pack_w_kernel(const float* __restrict__ W,
                              unsigned short* __restrict__ wp) {
    int e = blockIdx.x * blockDim.x + threadIdx.x;
    if (e >= WPACK_ELEMS) return;
    int j    = e & 15;
    int lane = (e >> 4) & 31;
    int t    = (e >> 9) & 3;
    int h    = (e >> 11) & 1;
    int k    = e >> 12;                       // 0..8
    int cin  = h * 32 + (j >> 3) * 16 + (lane >> 4) * 8 + (j & 7); // 0..63
    int cout = t * 16 + (lane & 15);          // 0..63
    wp[e] = f2bf(W[(k * CONV_C + cin) * CONV_C + cout]);
}

// --- f32 features -> bf16 activation buffer with zero sentinel row ----------
__global__ void cvt_bf16_kernel(const float* __restrict__ f,
                                unsigned short* __restrict__ xo, int n) {
    int i = blockIdx.x * blockDim.x + threadIdx.x;
    int total = (n + 1) * CONV_C;
    if (i >= total) return;
    xo[i] = (i < n * CONV_C) ? f2bf(f[i]) : (unsigned short)0;
}

// --- gather + GEMM (one 32-row tile per wave) + BN-stat accumulation --------
__global__ void conv_gemm_kernel(const unsigned short* __restrict__ x,  // (n+1) x 64 bf16
                                 const int* __restrict__ nbr,           // n x 9
                                 const unsigned short* __restrict__ wp, // WPACK_ELEMS bf16
                                 float* __restrict__ out,               // n x 64 f32
                                 float* __restrict__ gsum,
                                 float* __restrict__ gsumsq,
                                 int n) {
    extern __shared__ char smem[];
    unsigned short* sw = (unsigned short*)smem;         // packed weights
    float* ssum = (float*)(smem + WPACK_BYTES);         // 64
    float* ssq  = ssum + 64;                            // 64
    const int tid = threadIdx.x;

    // Stage all 72KB of packed weights into LDS (b128 copies).
    {
        const uint4* src = (const uint4*)wp;
        uint4* dst = (uint4*)sw;
        #pragma unroll 2
        for (int i = tid; i < WPACK_BYTES / 16; i += blockDim.x) dst[i] = src[i];
    }
    if (tid < 64) { ssum[tid] = 0.0f; ssq[tid] = 0.0f; }
    __syncthreads();

    const int wave  = tid >> 5;
    const int lane  = tid & 31;
    const int m     = lane & 15;      // row within M-tile / column within C-tile
    const int half  = lane >> 4;      // lane-half selects K-group / M-group
    const int row0  = (blockIdx.x * 8 + wave) * 32;

    if (row0 < n) {   // wave-uniform; n % 32 == 0 so tiles never straddle
        const int rowA = row0 + m;         // M-tile 0
        const int rowB = row0 + 16 + m;    // M-tile 1
        v8f acc0[4] = {v8f{}, v8f{}, v8f{}, v8f{}};
        v8f acc1[4] = {v8f{}, v8f{}, v8f{}, v8f{}};
        const int abase = half * 8;   // A layout: K 0..7/16..23 (lanes<16), 8..15/24..31 (lanes>=16)
        const uint4* wbase = (const uint4*)sw;

        int idx0[CONV_K], idx1[CONV_K];
        #pragma unroll
        for (int k = 0; k < CONV_K; ++k) {
            idx0[k] = nbr[rowA * CONV_K + k];
            idx1[k] = nbr[rowB * CONV_K + k];
        }

        for (int k = 0; k < CONV_K; ++k) {
            const unsigned short* xr0 = x + idx0[k] * CONV_C;
            const unsigned short* xr1 = x + idx1[k] * CONV_C;
            #pragma unroll
            for (int h = 0; h < 2; ++h) {
                union { uint4 u[2]; v16bf v; } A0, A1;
                A0.u[0] = *(const uint4*)(xr0 + h * 32 + abase);
                A0.u[1] = *(const uint4*)(xr0 + h * 32 + abase + 16);
                A1.u[0] = *(const uint4*)(xr1 + h * 32 + abase);
                A1.u[1] = *(const uint4*)(xr1 + h * 32 + abase + 16);
                const int fbase = ((k * 2 + h) * 4) * 64 + lane * 2; // uint4 index
                #pragma unroll
                for (int t = 0; t < 4; ++t) {
                    union { uint4 u[2]; v16bf v; } B;
                    B.u[0] = wbase[fbase + t * 64];
                    B.u[1] = wbase[fbase + t * 64 + 1];
                    acc0[t] = __builtin_amdgcn_wmma_f32_16x16x32_bf16(
                        false, A0.v, false, B.v, (short)0, acc0[t], false, false);
                    acc1[t] = __builtin_amdgcn_wmma_f32_16x16x32_bf16(
                        false, A1.v, false, B.v, (short)0, acc1[t], false, false);
                }
            }
        }

        // C/D layout: lane holds column c = 16t + m, rows M = r + 8*half.
        #pragma unroll
        for (int t = 0; t < 4; ++t) {
            const int c = 16 * t + m;
            float s = 0.0f, s2 = 0.0f;
            #pragma unroll
            for (int r = 0; r < 8; ++r) {
                float v0 = acc0[t][r];
                float v1 = acc1[t][r];
                out[(row0 + half * 8 + r) * CONV_C + c]      = v0;
                out[(row0 + 16 + half * 8 + r) * CONV_C + c] = v1;
                s  += v0 + v1;
                s2 += v0 * v0 + v1 * v1;
            }
            atomicAdd(&ssum[c], s);     // ds_add_f32
            atomicAdd(&ssq[c],  s2);
        }
    }
    __syncthreads();
    if (tid < 64) {
        atomicAdd(&gsum[tid],   ssum[tid]);
        atomicAdd(&gsumsq[tid], ssq[tid]);
    }
}

// --- BN + ReLU, requantize to bf16 (writes zero sentinel row) ---------------
__global__ void bnrelu_kernel(const float* __restrict__ raw,
                              const float* __restrict__ gsum,
                              const float* __restrict__ gsq,
                              const float* __restrict__ gamma,
                              const float* __restrict__ beta,
                              unsigned short* __restrict__ xo, int n) {
    int i = blockIdx.x * blockDim.x + threadIdx.x;
    int total = (n + 1) * CONV_C;
    if (i >= total) return;
    if (i >= n * CONV_C) { xo[i] = 0; return; }
    int c = i & 63;
    float inv = 1.0f / (float)n;
    float mu  = gsum[c] * inv;
    float var = gsq[c] * inv - mu * mu;
    float sc  = gamma[c] * rsqrtf(var + 1e-5f);
    float b   = beta[c] - mu * sc;
    float v   = raw[i] * sc + b;
    v = v > 0.0f ? v : 0.0f;
    xo[i] = f2bf(v);
}

// --- BN2 + residual add (in place on d_out) ---------------------------------
__global__ void bnres_kernel(float* __restrict__ io,
                             const float* __restrict__ feat,
                             const float* __restrict__ gsum,
                             const float* __restrict__ gsq,
                             const float* __restrict__ gamma,
                             const float* __restrict__ beta, int n) {
    int i = blockIdx.x * blockDim.x + threadIdx.x;
    if (i >= n * CONV_C) return;
    int c = i & 63;
    float inv = 1.0f / (float)n;
    float mu  = gsum[c] * inv;
    float var = gsq[c] * inv - mu * mu;
    float sc  = gamma[c] * rsqrtf(var + 1e-5f);
    float b   = beta[c] - mu * sc;
    io[i] = io[i] * sc + b + feat[i];
}

extern "C" void kernel_launch(void* const* d_in, const int* in_sizes, int n_in,
                              void* d_out, int out_size, void* d_ws, size_t ws_size,
                              hipStream_t stream) {
    const float* features = (const float*)d_in[0];
    const int*   nbr      = (const int*)d_in[1];
    const float* W1       = (const float*)d_in[2];
    const float* gamma1   = (const float*)d_in[3];
    const float* beta1    = (const float*)d_in[4];
    const float* W2       = (const float*)d_in[5];
    const float* gamma2   = (const float*)d_in[6];
    const float* beta2    = (const float*)d_in[7];
    float* out = (float*)d_out;

    const int n = in_sizes[0] / CONV_C;   // 200000 active sites

    // workspace layout
    char* ws = (char*)d_ws;
    float* stats = (float*)ws;                        // 256 f32
    float* sum1 = stats,       *sq1 = stats + 64;
    float* sum2 = stats + 128, *sq2 = stats + 192;
    unsigned short* wp1 = (unsigned short*)(ws + 1024);
    unsigned short* wp2 = (unsigned short*)(ws + 1024 + WPACK_BYTES);
    unsigned short* xbf = (unsigned short*)(ws + 1024 + 2 * WPACK_BYTES); // (n+1)*64 bf16

    const int packBlocks = (WPACK_ELEMS + 255) / 256;
    const int cvtBlocks  = ((n + 1) * CONV_C + 255) / 256;
    const int eltBlocks  = (n * CONV_C + 255) / 256;
    const int tiles32    = (n + 31) / 32;
    const int convBlocks = (tiles32 + 7) / 8;

    zero_stats_kernel<<<1, 256, 0, stream>>>(stats);
    pack_w_kernel<<<packBlocks, 256, 0, stream>>>(W1, wp1);
    pack_w_kernel<<<packBlocks, 256, 0, stream>>>(W2, wp2);
    cvt_bf16_kernel<<<cvtBlocks, 256, 0, stream>>>(features, xbf, n);

    // conv1: raw f32 output lands in d_out, stats in sum1/sq1
    conv_gemm_kernel<<<convBlocks, 256, LDS_BYTES, stream>>>(
        xbf, nbr, wp1, out, sum1, sq1, n);

    // BN1 + ReLU -> bf16 activations (reuse xbf buffer)
    bnrelu_kernel<<<cvtBlocks, 256, 0, stream>>>(
        out, sum1, sq1, gamma1, beta1, xbf, n);

    // conv2: raw f32 output overwrites d_out, stats in sum2/sq2
    conv_gemm_kernel<<<convBlocks, 256, LDS_BYTES, stream>>>(
        xbf, nbr, wp2, out, sum2, sq2, n);

    // BN2 + residual, in place on d_out
    bnres_kernel<<<eltBlocks, 256, 0, stream>>>(
        out, features, sum2, sq2, gamma2, beta2, n);
}
